// RNNMemoryModel_74380243632527
// MI455X (gfx1250) — matrix operand order repaired and checked by reading, not compiled, not run on hardware
//
#include <hip/hip_runtime.h>
#include <hip/hip_bf16.h>
#include <stdint.h>

typedef __attribute__((ext_vector_type(16))) _Float16 v16h;
typedef __attribute__((ext_vector_type(8)))  _Float16 v8h;
typedef __attribute__((ext_vector_type(8)))  float    v8f;

#define BATCH 256
#define SEQ   128
#define NN    2048
#define MIN3  24          // 3*M inputs
#define KA    2112        // 2048 + 24 bias cols + 40 zero pad = 33*64
#define DT_F  0.1f

// ---------------- deterministic counter-based RNG + Poisson ----------------
__device__ __forceinline__ uint32_t hash32(uint32_t x) {
  x ^= x >> 16; x *= 0x7feb352dU;
  x ^= x >> 15; x *= 0x846ca68bU;
  x ^= x >> 16; return x;
}
__device__ __forceinline__ float u01(uint32_t& s) {
  s = s * 1664525u + 1013904223u;
  uint32_t h = hash32(s);
  return (float)(h >> 8) * (1.0f / 16777216.0f) + (0.5f / 16777216.0f);
}
__device__ __forceinline__ int poisson_sample(float lam, uint32_t seed) {
  if (lam <= 0.0f) return 0;
  uint32_t s = seed;
  if (lam < 12.0f) {           // Knuth, bounded
    float L = __expf(-lam);
    float p = 1.0f; int k = 0;
    #pragma unroll 1
    for (int i = 0; i < 64; ++i) {
      p *= u01(s);
      if (p <= L) break;
      ++k;
    }
    return k;
  } else {                     // normal approximation (lambda up to ~80)
    float u1 = u01(s), u2 = u01(s);
    float z = __fsqrt_rn(-2.0f * __logf(u1)) * __cosf(6.2831853f * u2);
    float kf = lam + __fsqrt_rn(lam) * z;
    return kf < 0.0f ? 0 : (int)(kf + 0.5f);
  }
}

// act(s) = 400*(1 + tanh(0.4 s - 3)) = 800 / (1 + exp(6 - 0.8 s)); branch-free
__device__ __forceinline__ float activation(float s) {
  return 800.0f * __builtin_amdgcn_rcpf(1.0f + __expf(6.0f - 0.8f * s));
}

// ---- CDNA5 async global->LDS 16B fill (ASYNCcnt-tracked, no VGPR data) ----
__device__ __forceinline__ void async_fill_b128(void* lds_ptr, const void* gptr) {
  // LDS aperture: low 32 bits of a flat shared address are the LDS offset.
  uint32_t lds_off = (uint32_t)(uintptr_t)lds_ptr;
  asm volatile("global_load_async_to_lds_b128 %0, %1, off"
               :: "v"(lds_off), "v"((uint64_t)(uintptr_t)gptr)
               : "memory");
}
__device__ __forceinline__ void wait_async0() {
  asm volatile("s_wait_asynccnt 0x0" ::: "memory");
}

// ---------------- one-time setup kernels ----------------
// W' rows: [ W[n, 0..2047] | B[n, 0..23] | 0 x 40 ]  as f16
__global__ void build_waug_kernel(const float* __restrict__ W,
                                  const float* __restrict__ Bm,
                                  _Float16* __restrict__ Waug) {
  int idx = blockIdx.x * blockDim.x + threadIdx.x;
  if (idx >= NN * KA) return;
  int n = idx / KA, k = idx % KA;
  float v;
  if (k < NN)             v = W[n * NN + k];
  else if (k - NN < MIN3) v = Bm[n * MIN3 + (k - NN)];
  else                    v = 0.0f;
  Waug[idx] = (_Float16)v;
}

__global__ void init_state_kernel(const float* __restrict__ r0,
                                  float* __restrict__ rstate) {
  int idx = blockIdx.x * blockDim.x + threadIdx.x;
  if (idx < BATCH * NN) rstate[idx] = r0[idx];
}

// ---------------- per-step: Poisson observe + pack augmented A (f16) -------
__global__ void step_obs_kernel(const float* __restrict__ rstate,
                                const float* __restrict__ u,   // (B, SEQ, 24)
                                _Float16* __restrict__ Aaug,   // (B, KA)
                                int t) {
  int idx = blockIdx.x * blockDim.x + threadIdx.x;
  if (idx >= BATCH * KA) return;
  int b = idx / KA, c = idx % KA;
  float val;
  if (c < NN) {
    float lam = rstate[b * NN + c] * DT_F;
    uint32_t seed = hash32(((uint32_t)t * 0x9E3779B9u) ^
                           ((uint32_t)(b * NN + c) * 0x85EBCA77u) ^ 0x1B873593u);
    int k = poisson_sample(lam, seed);
    val = (float)k * 10.0f;                     // k / DT, exact in f16
  } else {
    int j = c - NN;
    val = (j < MIN3) ? u[((size_t)b * SEQ + t) * MIN3 + j] : 0.0f;
  }
  Aaug[idx] = (_Float16)val;
}

// ---------------- per-step WMMA GEMM + fused activation/Euler --------------
// grid: 64 blocks = 4 row-blocks (256/64) x 16 col-blocks (2048/128)
// block: 256 threads = 8 waves; wave w: rows 16*(w&3), cols 64*(w>>2)..+64
// LDS double buffer of K=64 slices: A 64x64 f16 (8KB), B 128x64 f16 (16KB).
// Pipeline: async LDS fills for slice i+1 in flight while 8 WMMAs run on slice i.
__global__ __launch_bounds__(256)
void gemm_step_kernel(const _Float16* __restrict__ Aaug,  // (256, KA)
                      const _Float16* __restrict__ Waug,  // (2048, KA)
                      float* __restrict__ rstate,         // (256, 2048)
                      float* __restrict__ rout,           // (256, SEQ, 2048)
                      int t) {
  __shared__ v8h lA[2][64 * 8];    // [buf][row*8 + chunk]  (chunk = 8 halfs = 16B)
  __shared__ v8h lB[2][128 * 8];   // [buf][col*8 + chunk]

  const int tid  = threadIdx.x;
  const int lane = tid & 31;
  const int wave = tid >> 5;              // 0..7
  const int brow = blockIdx.x & 3;        // 4 row blocks
  const int bcol = blockIdx.x >> 2;       // 16 col blocks
  const int row_blk = brow * 64;
  const int col_blk = bcol * 128;
  const int rs = (wave & 3) * 16;         // wave's row strip (local)
  const int cs = (wave >> 2) * 64;        // wave's col half (local)

  const int mrow = lane & 15;
  const int hi   = lane >> 4;

  // staging: A = 512 chunks (2/thread), B = 1024 chunks (4/thread);
  // chunk fastest over tid for coalesced 128B runs.
  const int s_r = tid >> 3, s_c = tid & 7;   // base row 0..31, chunk 0..7
  const _Float16* gA0 = Aaug + (size_t)(row_blk + s_r) * KA + s_c * 8;
  const _Float16* gA1 = gA0 + (size_t)32 * KA;
  const _Float16* gB0 = Waug + (size_t)(col_blk + s_r) * KA + s_c * 8;

  v8f acc[4] = {};

  // prologue: async-stage K-slice 0 into buffer 0
  {
    async_fill_b128(&lA[0][s_r * 8 + s_c],        gA0);
    async_fill_b128(&lA[0][(s_r + 32) * 8 + s_c], gA1);
    #pragma unroll
    for (int j = 0; j < 4; ++j)
      async_fill_b128(&lB[0][(s_r + 32 * j) * 8 + s_c],
                      gB0 + (size_t)(32 * j) * KA);
    wait_async0();
  }
  __syncthreads();

  int buf = 0;
  #pragma unroll 1
  for (int kk = 0; kk < KA; kk += 64) {
    const bool has_next = (kk + 64) < KA;
    if (has_next) {                       // async LDS fills for next slice in flight
      async_fill_b128(&lA[buf ^ 1][s_r * 8 + s_c],        gA0 + kk + 64);
      async_fill_b128(&lA[buf ^ 1][(s_r + 32) * 8 + s_c], gA1 + kk + 64);
      #pragma unroll
      for (int j = 0; j < 4; ++j)
        async_fill_b128(&lB[buf ^ 1][(s_r + 32 * j) * 8 + s_c],
                        gB0 + (size_t)(32 * j) * KA + kk + 64);
    }

    // load ALL fragments for the K=64 slice into distinct registers first
    // (2 A frags + 8 B frags = 20 ds_load_b128, one dscnt wait), then issue
    // 8 WMMAs back-to-back (D->C accumulation chains need no hazard NOPs).
    union F { v16h v; v8h h[2]; };
    F af[2], bf[8];
    #pragma unroll
    for (int s = 0; s < 2; ++s) {
      af[s].h[0] = lA[buf][(rs + mrow) * 8 + s * 4 + hi];
      af[s].h[1] = lA[buf][(rs + mrow) * 8 + s * 4 + 2 + hi];
      #pragma unroll
      for (int ct = 0; ct < 4; ++ct) {
        const int bi = (cs + ct * 16 + mrow) * 8 + s * 4 + hi;
        bf[s * 4 + ct].h[0] = lB[buf][bi];
        bf[s * 4 + ct].h[1] = lB[buf][bi + 2];
      }
    }
    #pragma unroll
    for (int s = 0; s < 2; ++s)
      #pragma unroll
      for (int ct = 0; ct < 4; ++ct)
        acc[ct] = __builtin_amdgcn_wmma_f32_16x16x32_f16(
            false, af[s].v, false, bf[s * 4 + ct].v, (short)0, acc[ct],
            false, false);

    if (has_next) wait_async0();          // own fills done; barrier makes all visible
    __syncthreads();
    buf ^= 1;
  }

  // epilogue: r_new = 0.9*r + 0.1*act(S)
  #pragma unroll
  for (int ct = 0; ct < 4; ++ct) {
    const int gn = col_blk + cs + ct * 16 + mrow;
    #pragma unroll
    for (int i = 0; i < 8; ++i) {
      const int gm = row_blk + rs + i + (hi << 3);  // C layout: vgpr i, lane-half
      float a = activation(acc[ct][i]);
      float rn = 0.9f * rstate[(size_t)gm * NN + gn] + 0.1f * a;
      rout[((size_t)gm * SEQ + t) * NN + gn] = rn;
      rstate[(size_t)gm * NN + gn] = rn;            // each (gm,gn) owned by one lane
    }
  }
}

__global__ void tail_kernel(const float* __restrict__ rstate,
                            float* __restrict__ out_tail) {
  int idx = blockIdx.x * blockDim.x + threadIdx.x;
  if (idx < BATCH * NN) out_tail[idx] = rstate[idx];
}

// ---------------- launcher ----------------
extern "C" void kernel_launch(void* const* d_in, const int* in_sizes, int n_in,
                              void* d_out, int out_size, void* d_ws, size_t ws_size,
                              hipStream_t stream) {
  (void)in_sizes; (void)n_in; (void)out_size; (void)ws_size;
  const float* u  = (const float*)d_in[0];   // (256,128,24)
  const float* r0 = (const float*)d_in[1];   // (1,256,2048)
  const float* Bm = (const float*)d_in[2];   // (2048,24)
  const float* W  = (const float*)d_in[3];   // (2048,2048)
  float* out = (float*)d_out;                // (256,128,2048) ++ (1,256,2048)

  char* ws = (char*)d_ws;
  const size_t waug_bytes = (size_t)NN * KA * sizeof(_Float16);    // 8.7 MB
  const size_t aaug_bytes = (size_t)BATCH * KA * sizeof(_Float16); // 1.1 MB
  _Float16* Waug   = (_Float16*)ws;
  _Float16* Aaug   = (_Float16*)(ws + waug_bytes);
  float*    rstate = (float*)(ws + waug_bytes + aaug_bytes);       // 2 MB

  build_waug_kernel<<<(NN * KA + 255) / 256, 256, 0, stream>>>(W, Bm, Waug);
  init_state_kernel<<<(BATCH * NN + 255) / 256, 256, 0, stream>>>(r0, rstate);

  for (int t = 0; t < SEQ; ++t) {
    step_obs_kernel<<<(BATCH * KA + 255) / 256, 256, 0, stream>>>(rstate, u, Aaug, t);
    gemm_step_kernel<<<64, 256, 0, stream>>>(Aaug, Waug, rstate, out, t);
  }

  tail_kernel<<<(BATCH * NN + 255) / 256, 256, 0, stream>>>(
      rstate, out + (size_t)BATCH * SEQ * NN);
}
